// SpatialAttentionLayer_24412594111213
// MI455X (gfx1250) — compile-verified, matrix operand order checked by reading, not run on hardware
//
#include <hip/hip_runtime.h>
#include <math.h>

// ---------------------------------------------------------------------------
// SpatialAttentionLayer for MI455X (gfx1250, wave32, WMMA bf16 + async LDS)
// out = softmax_one( (Q K^T / 16) * exp(-d2/2) ) @ V + Xf
// ---------------------------------------------------------------------------

typedef __attribute__((ext_vector_type(16))) __bf16 v16bf;
typedef __attribute__((ext_vector_type(8)))  __bf16 v8bf;
typedef __attribute__((ext_vector_type(8)))  float  v8f;

#define NROWS 8192
#define DIM   256
#define XLD   259            // 256 features + 3 position cols
#define SPLITS 4
#define KEYS_PER_SPLIT (NROWS / SPLITS)
#define TILE_KEYS 64
#define NTILES (KEYS_PER_SPLIT / TILE_KEYS)
#define KROW   264           // 256 halves + 8 pad  -> 528B row, bank-staggered
#define AROW   88            // 64 halves + 24 pad  -> 176B row, bank-staggered
#define NEG_BIG (-3.0e38f)

static __device__ __forceinline__ v16bf load_op16(const __bf16* p0, const __bf16* p1) {
    v16bf r;
    *(v8bf*)&r       = *(const v8bf*)p0;   // 16B -> global_load_b128 / ds_load_b128
    *((v8bf*)&r + 1) = *(const v8bf*)p1;   // 16B
    return r;
}

static __device__ __forceinline__ v8f zero8() {
    v8f z;
#pragma unroll
    for (int i = 0; i < 8; ++i) z[i] = 0.0f;
    return z;
}

// CDNA5 async global->LDS copy (ASYNCcnt-tracked), 16B per lane.
static __device__ __forceinline__ void async_copy16(unsigned int lds_off, const void* gaddr) {
    asm volatile("global_load_async_to_lds_b128 %0, %1, off"
                 :: "v"(lds_off), "v"(gaddr) : "memory");
}

static __device__ __forceinline__ void wait_async0() {
#if __has_builtin(__builtin_amdgcn_s_wait_asynccnt)
    __builtin_amdgcn_s_wait_asynccnt(0);
#else
    asm volatile("s_wait_asynccnt 0" ::: "memory");
#endif
}

static __device__ __forceinline__ void wait_ds0() {
    asm volatile("s_wait_dscnt 0" ::: "memory");
}

// ---------------------------------------------------------------------------
// Phase 0a: X (8192 x 259 f32) -> Xbf (8192 x 256 bf16), P+|P|^2 -> float4
// ---------------------------------------------------------------------------
__global__ void k_convX(const float* __restrict__ X,
                        __bf16* __restrict__ Xbf,
                        float4* __restrict__ P) {
    int t   = threadIdx.x;
    int row = blockIdx.x * 4 + (t >> 6);
    int c   = (t & 63) * 4;
    const float* xr = X + (size_t)row * XLD;
    __bf16* o = Xbf + (size_t)row * DIM + c;
    o[0] = (__bf16)xr[c + 0];
    o[1] = (__bf16)xr[c + 1];
    o[2] = (__bf16)xr[c + 2];
    o[3] = (__bf16)xr[c + 3];
    if ((t & 63) == 0) {
        float p0 = xr[256], p1 = xr[257], p2 = xr[258];
        P[row] = make_float4(p0, p1, p2, p0 * p0 + p1 * p1 + p2 * p2);
    }
}

// ---------------------------------------------------------------------------
// Phase 0b: W (256x256 f32, k-major) -> Wt (bf16, n-major) so WMMA B-operand
// k-pairs are contiguous.
// ---------------------------------------------------------------------------
__global__ void k_convW(const float* __restrict__ WQ,
                        const float* __restrict__ WK,
                        const float* __restrict__ WV,
                        __bf16* __restrict__ Wt) {
    int idx = blockIdx.x * 256 + threadIdx.x;
    int mat = idx >> 16;
    int rem = idx & 65535;
    int n = rem >> 8, k = rem & 255;
    const float* W = (mat == 0) ? WQ : (mat == 1) ? WK : WV;
    Wt[(size_t)mat * 65536 + (size_t)n * 256 + k] = (__bf16)W[(size_t)k * 256 + n];
}

// ---------------------------------------------------------------------------
// Phase 1: Q = Xf WQ + bQ ; K = Xf WK + bK ; V = Xf WV + bV   (bf16 out)
// Q,K row-major [row][feat]; V transposed [feat][row].
// ---------------------------------------------------------------------------
__global__ __launch_bounds__(128) void k_qkv(
        const __bf16* __restrict__ Xbf, const __bf16* __restrict__ Wt,
        const float* __restrict__ bQ, const float* __restrict__ bK,
        const float* __restrict__ bV,
        __bf16* __restrict__ Qb, __bf16* __restrict__ Kb, __bf16* __restrict__ Vt) {
    int lane = threadIdx.x & 31, w = threadIdx.x >> 5;
    int ln = lane & 15, hf = lane >> 4;
    int mat = blockIdx.y >> 2;
    int cb  = (blockIdx.y & 3) * 64;
    int rb  = blockIdx.x * 64 + w * 16;
    const float*  bias = (mat == 0) ? bQ : (mat == 1) ? bK : bV;
    const __bf16* Wm   = Wt + (size_t)mat * 65536;

    v8f acc[4];
#pragma unroll
    for (int nt = 0; nt < 4; ++nt) {
        float bv = bias[cb + nt * 16 + ln];
#pragma unroll
        for (int i = 0; i < 8; ++i) acc[nt][i] = bv;
    }
#pragma unroll
    for (int ks = 0; ks < 8; ++ks) {
        int koff = ks * 32 + hf * 8;
        const __bf16* ap = Xbf + (size_t)(rb + ln) * DIM + koff;
        v16bf aop = load_op16(ap, ap + 16);
#pragma unroll
        for (int nt = 0; nt < 4; ++nt) {
            const __bf16* bp = Wm + (size_t)(cb + nt * 16 + ln) * DIM + koff;
            v16bf bop = load_op16(bp, bp + 16);
            acc[nt] = __builtin_amdgcn_wmma_f32_16x16x32_bf16(
                false, aop, false, bop, (short)0, acc[nt], false, false);
        }
    }
#pragma unroll
    for (int nt = 0; nt < 4; ++nt) {
        int col = cb + nt * 16 + ln;
#pragma unroll
        for (int r = 0; r < 8; ++r) {
            int row = rb + hf * 8 + r;
            __bf16 val = (__bf16)acc[nt][r];
            if (mat == 0)      Qb[(size_t)row * DIM + col] = val;
            else if (mat == 1) Kb[(size_t)row * DIM + col] = val;
            else               Vt[(size_t)col * NROWS + row] = val;   // transposed
        }
    }
}

// ---------------------------------------------------------------------------
// Phase 2: flash attention, split over keys. Block = 4 waves x 16 query rows.
// K tile (64x256 bf16) staged in LDS via async copies; copy of tile t+1
// overlaps softmax + A@V of tile t. Emits partial O, row max M, row sum L.
// ---------------------------------------------------------------------------
__global__ __launch_bounds__(128) void k_attn(
        const __bf16* __restrict__ Qb, const __bf16* __restrict__ Kb,
        const __bf16* __restrict__ Vt, const float4* __restrict__ P,
        float* __restrict__ Opart, float* __restrict__ Mbuf,
        float* __restrict__ Lbuf) {
    __shared__ __align__(16) __bf16 Ktile[TILE_KEYS][KROW];   // 33 KB, padded rows
    __shared__ __align__(16) __bf16 aStage[4][16][AROW];      // 11 KB, padded rows
    __shared__ float4 qP[64];

    int t = threadIdx.x;
    int lane = t & 31, w = t >> 5;
    int ln = lane & 15, hf = lane >> 4;
    int qb = blockIdx.x * 64 + w * 16;
    int split = blockIdx.y;
    int kstart = split * KEYS_PER_SPLIT;

    unsigned int ldsK = (unsigned int)(size_t)&Ktile[0][0];   // LDS byte offset

    // Async-stage one 64x256 bf16 K tile: 32KB contiguous in global,
    // 528B-padded rows in LDS. 16 x b128 per thread.
    auto issue_tile = [&](int kbase) {
        const char* gk = (const char*)(Kb + (size_t)kbase * DIM);
#pragma unroll
        for (int i = 0; i < 16; ++i) {
            int o   = (i * 128 + t) * 16;          // linear byte offset in tile
            int row = o >> 9;                      // / 512
            int col = o & 511;
            async_copy16(ldsK + (unsigned int)(row * (KROW * 2) + col), gk + o);
        }
    };

    issue_tile(kstart);                                        // prologue copy

    if (hf == 0) qP[w * 16 + ln] = P[qb + ln];
    __builtin_prefetch(Qb + (size_t)(qb + ln) * DIM, 0, 3);    // global_prefetch_b8

    v8f O[16];
#pragma unroll
    for (int i = 0; i < 16; ++i) O[i] = zero8();
    float Mrun[8], Lrun[8];
#pragma unroll
    for (int r = 0; r < 8; ++r) { Mrun[r] = NEG_BIG; Lrun[r] = 0.0f; }

    wait_async0();
    __syncthreads();

    for (int kt = 0; kt < NTILES; ++kt) {
        int kbase = kstart + kt * TILE_KEYS;

        float4 kp[4];
#pragma unroll
        for (int nt = 0; nt < 4; ++nt) kp[nt] = P[kbase + nt * 16 + ln];

        // ---- S = Q @ K^T from LDS K tile ----------------------------------
        v8f acc[4];
#pragma unroll
        for (int nt = 0; nt < 4; ++nt) acc[nt] = zero8();
#pragma unroll
        for (int ks = 0; ks < 8; ++ks) {
            int koff = ks * 32 + hf * 8;
            const __bf16* qp_ = Qb + (size_t)(qb + ln) * DIM + koff;
            v16bf aop = load_op16(qp_, qp_ + 16);
#pragma unroll
            for (int nt = 0; nt < 4; ++nt) {
                const __bf16* bp = &Ktile[nt * 16 + ln][koff];
                v16bf bop = load_op16(bp, bp + 16);     // ds_load_b128 x2
                acc[nt] = __builtin_amdgcn_wmma_f32_16x16x32_bf16(
                    false, aop, false, bop, (short)0, acc[nt], false, false);
            }
        }

        // All waves done reading Ktile -> safe to overwrite with next tile.
        __syncthreads();
        if (kt + 1 < NTILES) issue_tile(kbase + TILE_KEYS);   // overlapped copy

        // ---- decay * scores, online softmax_one ---------------------------
        float oscale[8];
#pragma unroll
        for (int r = 0; r < 8; ++r) {
            float4 q4 = qP[w * 16 + hf * 8 + r];
            float mx = NEG_BIG;
#pragma unroll
            for (int nt = 0; nt < 4; ++nt) {
                float sc = acc[nt][r] * 0.0625f;       // 1/sqrt(256)
                float4 k4 = kp[nt];
                float d2 = q4.w + k4.w - 2.0f * (q4.x * k4.x + q4.y * k4.y + q4.z * k4.z);
                d2 = fmaxf(d2, 0.0f);
                float x = sc * __expf(-0.5f * d2);
                acc[nt][r] = x;
                mx = fmaxf(mx, x);
            }
            mx = fmaxf(mx, __shfl_xor(mx, 1, 16));
            mx = fmaxf(mx, __shfl_xor(mx, 2, 16));
            mx = fmaxf(mx, __shfl_xor(mx, 4, 16));
            mx = fmaxf(mx, __shfl_xor(mx, 8, 16));
            float Mn   = fmaxf(Mrun[r], mx);
            float corr = __expf(Mrun[r] - Mn);
            Mrun[r] = Mn;
            float sum = 0.0f;
#pragma unroll
            for (int nt = 0; nt < 4; ++nt) {
                float p = __expf(acc[nt][r] - Mn);
                acc[nt][r] = p;
                sum += p;
            }
            sum += __shfl_xor(sum, 1, 16);
            sum += __shfl_xor(sum, 2, 16);
            sum += __shfl_xor(sum, 4, 16);
            sum += __shfl_xor(sum, 8, 16);
            Lrun[r] = Lrun[r] * corr + sum;
            oscale[r] = corr;
        }
#pragma unroll
        for (int ot = 0; ot < 16; ++ot)
#pragma unroll
            for (int r = 0; r < 8; ++r) O[ot][r] *= oscale[r];

        // ---- D-layout -> A-layout via per-wave LDS transpose --------------
#pragma unroll
        for (int nt = 0; nt < 4; ++nt)
#pragma unroll
            for (int r = 0; r < 8; ++r)
                aStage[w][hf * 8 + r][nt * 16 + ln] = (__bf16)acc[nt][r];
        wait_ds0();

        v16bf aop2[2];
#pragma unroll
        for (int kk = 0; kk < 2; ++kk) {
            const __bf16* ap = &aStage[w][ln][kk * 32 + hf * 8];
            aop2[kk] = load_op16(ap, ap + 16);
        }

        // ---- O += A @ V (V^T streamed from global/L2) ----------------------
#pragma unroll
        for (int ot = 0; ot < 16; ++ot) {
#pragma unroll
            for (int kk = 0; kk < 2; ++kk) {
                const __bf16* vp = Vt + (size_t)(ot * 16 + ln) * NROWS
                                   + kbase + kk * 32 + hf * 8;
                v16bf vop = load_op16(vp, vp + 16);
                O[ot] = __builtin_amdgcn_wmma_f32_16x16x32_bf16(
                    false, aop2[kk], false, vop, (short)0, O[ot], false, false);
            }
        }
        wait_ds0();        // WAR guard on aStage

        wait_async0();     // next K tile landed
        __syncthreads();
    }

    // ---- write split partials ---------------------------------------------
    float* Op = Opart + (size_t)split * NROWS * DIM;
#pragma unroll
    for (int ot = 0; ot < 16; ++ot) {
        int col = ot * 16 + ln;
#pragma unroll
        for (int r = 0; r < 8; ++r) {
            int row = qb + hf * 8 + r;
            Op[(size_t)row * DIM + col] = O[ot][r];
        }
    }
    if (ln == 0) {
#pragma unroll
        for (int r = 0; r < 8; ++r) {
            int row = qb + hf * 8 + r;
            Mbuf[(size_t)split * NROWS + row] = Mrun[r];
            Lbuf[(size_t)split * NROWS + row] = Lrun[r];
        }
    }
}

// ---------------------------------------------------------------------------
// Phase 3: merge split partials; denom = 1 + sum_s L_s e^{M_s-M}; add Xf.
// ---------------------------------------------------------------------------
__global__ void k_combine(const float* __restrict__ Opart,
                          const float* __restrict__ Mbuf,
                          const float* __restrict__ Lbuf,
                          const float* __restrict__ X,
                          float* __restrict__ out) {
    int idx = blockIdx.x * 256 + threadIdx.x;
    int row = idx >> 8;
    int col = idx & 255;
    float M = NEG_BIG;
#pragma unroll
    for (int s = 0; s < SPLITS; ++s) M = fmaxf(M, Mbuf[s * NROWS + row]);
    float denom = 1.0f, num = 0.0f;
#pragma unroll
    for (int s = 0; s < SPLITS; ++s) {
        float sc = __expf(Mbuf[s * NROWS + row] - M);
        denom += Lbuf[s * NROWS + row] * sc;
        num += Opart[(size_t)s * NROWS * DIM + (size_t)row * DIM + col] * sc;
    }
    out[(size_t)row * DIM + col] = num / denom + X[(size_t)row * XLD + col];
}

// ---------------------------------------------------------------------------
extern "C" void kernel_launch(void* const* d_in, const int* in_sizes, int n_in,
                              void* d_out, int out_size, void* d_ws, size_t ws_size,
                              hipStream_t stream) {
    (void)in_sizes; (void)n_in; (void)out_size; (void)ws_size;
    const float* X  = (const float*)d_in[0];
    const float* WQ = (const float*)d_in[1];
    const float* bQ = (const float*)d_in[2];
    const float* WK = (const float*)d_in[3];
    const float* bK = (const float*)d_in[4];
    const float* WV = (const float*)d_in[5];
    const float* bV = (const float*)d_in[6];
    float* out = (float*)d_out;

    char* ws = (char*)d_ws;
    size_t off = 0;
    __bf16* Xbf = (__bf16*)(ws + off); off += (size_t)NROWS * DIM * 2;          // 4 MB
    __bf16* Qb  = (__bf16*)(ws + off); off += (size_t)NROWS * DIM * 2;          // 4 MB
    __bf16* Kb  = (__bf16*)(ws + off); off += (size_t)NROWS * DIM * 2;          // 4 MB
    __bf16* Vt  = (__bf16*)(ws + off); off += (size_t)NROWS * DIM * 2;          // 4 MB
    __bf16* Wt  = (__bf16*)(ws + off); off += (size_t)3 * DIM * DIM * 2;        // 384 KB
    float4* P   = (float4*)(ws + off); off += (size_t)NROWS * 16;               // 128 KB
    float* Opart = (float*)(ws + off); off += (size_t)SPLITS * NROWS * DIM * 4; // 32 MB
    float* Mbuf  = (float*)(ws + off); off += (size_t)SPLITS * NROWS * 4;
    float* Lbuf  = (float*)(ws + off); off += (size_t)SPLITS * NROWS * 4;

    k_convX<<<NROWS / 4, 256, 0, stream>>>(X, Xbf, P);
    k_convW<<<(3 * DIM * DIM) / 256, 256, 0, stream>>>(WQ, WK, WV, Wt);
    k_qkv<<<dim3(NROWS / 64, 12), 128, 0, stream>>>(Xbf, Wt, bQ, bK, bV, Qb, Kb, Vt);
    k_attn<<<dim3(NROWS / 64, SPLITS), 128, 0, stream>>>(Qb, Kb, Vt, P, Opart, Mbuf, Lbuf);
    k_combine<<<(NROWS * DIM) / 256, 256, 0, stream>>>(Opart, Mbuf, Lbuf, X, out);
}
